// HDRPlusAlign_76304388980745
// MI455X (gfx1250) — matrix operand order repaired and checked by reading, not compiled, not run on hardware
//
#include <hip/hip_runtime.h>

typedef _Float16 f16;
typedef __attribute__((ext_vector_type(16))) _Float16 v16h;
typedef __attribute__((ext_vector_type(8)))  float    v8f;

#define T    16
#define SR   4
#define PADD 1
#define SS   11   // 2*(SR+PADD)+1
#define SSC  9    // 2*SR+1
#define WIN  26   // T + 2*(SR+PADD)
#define SW   0.1f
#define WAVES 8
#define KSTR 32   // K enumeration row stride (xw 26..31 are zero columns)
#define KCH  16   // K = 16*32 = 512 = 16 chunks of 32
#define QLEN 576  // 16*32 + pads

// ---------------------------------------------------------------------------
// 5x5 gaussian blur (zero pad) + 2x2 average pool, one thread per output px.
// ---------------------------------------------------------------------------
__global__ void gauss_down2(const float* __restrict__ in, float* __restrict__ out,
                            int n, int H, int W) {
    int H2 = H >> 1, W2 = W >> 1;
    int total = n * H2 * W2;
    int i = blockIdx.x * blockDim.x + threadIdx.x;
    if (i >= total) return;
    int x2 = i % W2, y2 = (i / W2) % H2, img = i / (W2 * H2);

    const float gs[5] = {0.13533528323661270f, 0.60653065971263342f, 1.0f,
                         0.60653065971263342f, 0.13533528323661270f};
    const float S = 2.0f * (0.13533528323661270f + 0.60653065971263342f) + 1.0f;
    const float inv = 1.0f / (S * S);

    const float* p = in + (size_t)img * H * W;
    float acc = 0.f;
    for (int q = 0; q < 2; ++q) {
        for (int r = 0; r < 2; ++r) {
            int yc = 2 * y2 + q, xc = 2 * x2 + r;
            float b = 0.f;
            for (int iy = -2; iy <= 2; ++iy) {
                int yy = yc + iy;
                if (yy < 0 || yy >= H) continue;   // zero pad
                float rowv = 0.f;
                for (int ix = -2; ix <= 2; ++ix) {
                    int xx = xc + ix;
                    if (xx < 0 || xx >= W) continue;
                    rowv += gs[ix + 2] * p[yy * W + xx];
                }
                b += gs[iy + 2] * rowv;
            }
            acc += b * inv;
        }
    }
    out[(size_t)img * H2 * W2 + y2 * W2 + x2] = acc * 0.25f;
}

// ---------------------------------------------------------------------------
// One pyramid level of block matching. One wave32 per 16x16 tile.
//   dist(dy,dx) = ( box2 - 2*cross + srcss ) / 256 + SW * sdist
// Cross term as one chained WMMA GEMM:
//   D(dy,dx) = sum_k win(y+dy, xw) * src(y, xw-dx),  k = y*32 + xw
//   M = dy (11 of 16), N = dx (11 of 16), K = 512 = 16 x 32.
// A: winh f16, stride 32 -> A(m,k) = winh[k + 32*m]          (linear in k)
// B: q    f16, zero-padded shifted tile -> B(k,n) = q[16+k-n] (linear in k)
// ---------------------------------------------------------------------------
struct WaveLds {
    float win[WIN][WIN + 2];   // f32 window (for exact box2 term)
    float colsum[SS][WIN];     // sliding 16-row column sums of win^2
    float dist[SS * SS];       // 11x11 dist_full
    f16   winh[31][KSTR];      // f16 window; cols 26..31 zero (rows 0..25)
    f16   q[QLEN];             // f16 tile embedded with zeros: q[16+y*32+x]
};

__global__ void __launch_bounds__(WAVES * 32)
align_step(const float* __restrict__ srcL, const float* __restrict__ dstL,
           const float* __restrict__ prevOff, float* __restrict__ outOff,
           float* __restrict__ outDist,
           int n, int H, int W, int nth, int ntw, int nthP, int ntwP) {
    __shared__ WaveLds L[WAVES];
    const int wave = threadIdx.x >> 5;
    const int lane = threadIdx.x & 31;
    WaveLds& S = L[wave];

    const int total = n * nth * ntw;
    int tl = blockIdx.x * WAVES + wave;
    const bool valid = tl < total;
    if (!valid) tl = 0;                       // keep all waves converged
    const int tx  = tl % ntw;
    const int ty  = (tl / ntw) % nth;
    const int img = tl / (ntw * nth);

    // ---- inherit initial offset from previous (coarser) level -------------
    float ioy = 0.f, iox = 0.f;
    if (prevOff) {
        const float* pp = prevOff +
            (((size_t)img * nthP + (ty >> 1)) * ntwP + (tx >> 1)) * 2;
        float oy2 = 2.f * pp[0], ox2 = 2.f * pp[1];
        float by = (float)(ty * T), bx = (float)(tx * T);
        ioy = rintf(fminf(fmaxf(oy2 + by, 0.f), (float)(H - T)) - by);
        iox = rintf(fminf(fmaxf(ox2 + bx, 0.f), (float)(W - T)) - bx);
    }
    const int oy = (int)ioy, ox = (int)iox;
    const int y0 = ty * T + oy - (SR + PADD);
    const int x0 = tx * T + ox - (SR + PADD);

    // ---- stage window (f32 + f16, edge-clamped) ---------------------------
    const float* dimg = dstL + (size_t)img * H * W;
    const float* simg = srcL + (size_t)img * H * W;
    for (int i = lane; i < WIN * KSTR; i += 32) {   // 26 rows x 32 cols
        int wy = i >> 5, wx = i & 31;
        float v = 0.f;
        if (wx < WIN) {
            int gy = min(max(y0 + wy, 0), H - 1);
            int gx = min(max(x0 + wx, 0), W - 1);
            v = dimg[gy * W + gx];
            S.win[wy][wx] = v;
        }
        S.winh[wy][wx] = (f16)v;                    // cols 26..31 -> 0
    }
    // ---- stage shifted/zero-padded src tile + Sum src^2 -------------------
    for (int i = lane; i < QLEN; i += 32) S.q[i] = (f16)0.f;
    float ss = 0.f;
    for (int i = lane; i < T * T; i += 32) {
        int yy = i >> 4, xx = i & 15;
        float v = simg[(ty * T + yy) * W + tx * T + xx];
        ss += v * v;
        S.q[16 + yy * KSTR + xx] = (f16)v;          // per-wave LDS is in-order
    }
    #pragma unroll
    for (int o = 16; o; o >>= 1) ss += __shfl_xor(ss, o, 32);
    const float srcss = ss;
    __syncthreads();

    // ---- sliding box sums of win^2 (Sum cand^2 term) ----------------------
    for (int i = lane; i < SS * WIN; i += 32) {
        int dy = i / WIN, x = i % WIN;
        float s = 0.f;
        #pragma unroll
        for (int y = 0; y < T; ++y) { float v = S.win[dy + y][x]; s += v * v; }
        S.colsum[dy][x] = s;
    }
    __syncthreads();
    for (int i = lane; i < SS * SS; i += 32) {
        int dy = i / SS, dx = i % SS;
        float s = 0.f;
        #pragma unroll
        for (int x = 0; x < T; ++x) s += S.colsum[dy][dx + x];
        S.dist[i] = s;                 // starts as box2
    }
    __syncthreads();

    // ---- cross term: single chained WMMA GEMM -----------------------------
    const int half = lane >> 4;        // lane-half selects K sub-range
    const int mn   = lane & 15;        // A row m=dy ; B col n=dx ; D: N
    const f16* pa = &S.winh[0][0] + mn * KSTR + 8 * half;
    const f16* pb = &S.q[0] + 16 + 8 * half - mn;
    v8f acc = {};
    #pragma unroll
    for (int kc = 0; kc < KCH; ++kc) {
        v16h a, b;
        #pragma unroll
        for (int e = 0; e < 8; ++e) {              // A K-slots: 8h+e, 16+8h+e
            a[e]     = pa[kc * KSTR + e];
            a[e + 8] = pa[kc * KSTR + 16 + e];
        }
        #pragma unroll
        for (int e = 0; e < 8; ++e) {              // B K-slots: 8h+e, 16+8h+e
            b[e]     = pb[kc * KSTR + e];
            b[e + 8] = pb[kc * KSTR + 16 + e];
        }
        acc = __builtin_amdgcn_wmma_f32_16x16x32_f16(
            false, a, false, b, (short)0, acc, false, false);
    }
    // D layout: lane holds N=mn (dx), rows M = r + 8*half (dy)
    #pragma unroll
    for (int r = 0; r < 8; ++r) {
        int dy = r + 8 * half;
        if (dy < SS && mn < SS)
            S.dist[dy * SS + mn] -= 2.f * acc[r];
    }
    __syncthreads();

    // ---- finalize dist_full ----------------------------------------------
    for (int i = lane; i < SS * SS; i += 32) {
        int dy = i / SS, dx = i % SS;
        float ay = (float)(dy - (SR + PADD)) / (float)SS;
        float ax = (float)(dx - (SR + PADD)) / (float)SS;
        S.dist[i] = (S.dist[i] + srcss) * (1.0f / (T * T))
                  + SW * (ay * ay + ax * ax);
    }
    __syncthreads();

    // ---- argmin over inner 9x9 (first-occurrence tie break) ---------------
    float bv = 3.4e38f; int bi = 1 << 20;
    for (int i = lane; i < SSC * SSC; i += 32) {
        int dy = i / SSC + PADD, dx = i % SSC + PADD;
        float v = S.dist[dy * SS + dx];
        if (v < bv) { bv = v; bi = i; }
    }
    #pragma unroll
    for (int o = 16; o; o >>= 1) {
        float ov = __shfl_xor(bv, o, 32);
        int   oi = __shfl_xor(bi, o, 32);
        if (ov < bv || (ov == bv && oi < bi)) { bv = ov; bi = oi; }
    }

    // ---- subpixel refine + store (lane 0) ---------------------------------
    if (lane == 0 && valid) {
        int py = bi / SSC, px = bi % SSC;
        float offy = ioy + (float)(py - SR);
        float offx = iox + (float)(px - SR);
        float yreg[3][3];
        for (int r = 0; r < 3; ++r)
            for (int c = 0; c < 3; ++c)
                yreg[r][c] = S.dist[(py + r) * SS + (px + c)];
        const float FA11[3][3] = {{1,-2,1},{2,-4,2},{1,-2,1}};
        const float FA22[3][3] = {{1,2,1},{-2,-4,-2},{1,2,1}};
        const float FA12[3][3] = {{1,0,-1},{0,0,0},{-1,0,1}};
        const float FB1 [3][3] = {{-1,0,1},{-2,0,2},{-1,0,1}};
        const float FB2 [3][3] = {{-1,-2,-1},{0,0,0},{1,2,1}};
        float a11=0,a22=0,a12=0,b1=0,b2=0;
        for (int r = 0; r < 3; ++r)
            for (int c = 0; c < 3; ++c) {
                float v = yreg[r][c];
                a11 += FA11[r][c]*v; a22 += FA22[r][c]*v; a12 += FA12[r][c]*v;
                b1  += FB1 [r][c]*v; b2  += FB2 [r][c]*v;
            }
        a11 = fmaxf(a11 * 0.25f, 0.f);
        a22 = fmaxf(a22 * 0.25f, 0.f);
        a12 *= 0.25f; b1 *= 0.125f; b2 *= 0.125f;
        float det  = a11 * a22 - a12 * a12;
        float a12z = (det < 0.f) ? 0.f : a12;
        float mux = -(a22 * b1 - a12z * b2) / det;
        float muy = -(a11 * b2 - a12z * b1) / det;
        float len = sqrtf(muy * muy + mux * mux);
        offy += (len < 1.f) ? muy : 0.f;     // NaN len -> false (matches jnp.where)
        offx += (len < 1.f) ? mux : 0.f;
        float* po = outOff + (size_t)tl * 2;
        po[0] = offy; po[1] = offx;
        if (outDist) outDist[tl] = bv;
    }
}

// ---------------------------------------------------------------------------
// Broadcast tile results to per-pixel outputs: [off (n,512,512,2) | dist].
// ---------------------------------------------------------------------------
__global__ void expand_out(const float* __restrict__ off0,
                           const float* __restrict__ dist0,
                           float* __restrict__ out,
                           int n, int Hp, int Wp, int nth, int ntw) {
    int total = n * Hp * Wp;
    int i = blockIdx.x * blockDim.x + threadIdx.x;
    if (i >= total) return;
    int x = i % Wp, y = (i / Wp) % Hp, img = i / (Wp * Hp);
    int tl = (img * nth + (y >> 4)) * ntw + (x >> 4);
    out[(size_t)i * 2 + 0]      = off0[tl * 2 + 0];
    out[(size_t)i * 2 + 1]      = off0[tl * 2 + 1];
    out[(size_t)total * 2 + i]  = dist0[tl];
}

static inline int cdiv(int a, int b) { return (a + b - 1) / b; }

extern "C" void kernel_launch(void* const* d_in, const int* in_sizes, int n_in,
                              void* d_out, int out_size, void* d_ws, size_t ws_size,
                              hipStream_t stream) {
    const float* src = (const float*)d_in[0];
    const float* dst = (const float*)d_in[1];
    float* out = (float*)d_out;
    float* ws  = (float*)d_ws;

    const int n = in_sizes[0] / (512 * 512);   // 4

    size_t o = 0;
    float* s1 = ws + o; o += (size_t)n * 256 * 256;
    float* s2 = ws + o; o += (size_t)n * 128 * 128;
    float* s3 = ws + o; o += (size_t)n * 64 * 64;
    float* t1 = ws + o; o += (size_t)n * 256 * 256;
    float* t2 = ws + o; o += (size_t)n * 128 * 128;
    float* t3 = ws + o; o += (size_t)n * 64 * 64;
    float* off3 = ws + o; o += (size_t)n * 4 * 4 * 2;
    float* off2 = ws + o; o += (size_t)n * 8 * 8 * 2;
    float* off1 = ws + o; o += (size_t)n * 16 * 16 * 2;
    float* off0 = ws + o; o += (size_t)n * 32 * 32 * 2;
    float* dist0 = ws + o; o += (size_t)n * 32 * 32;

    // Gaussian pyramids (src, dst)
    gauss_down2<<<cdiv(n*256*256,256),256,0,stream>>>(src, s1, n, 512, 512);
    gauss_down2<<<cdiv(n*128*128,256),256,0,stream>>>(s1,  s2, n, 256, 256);
    gauss_down2<<<cdiv(n* 64* 64,256),256,0,stream>>>(s2,  s3, n, 128, 128);
    gauss_down2<<<cdiv(n*256*256,256),256,0,stream>>>(dst, t1, n, 512, 512);
    gauss_down2<<<cdiv(n*128*128,256),256,0,stream>>>(t1,  t2, n, 256, 256);
    gauss_down2<<<cdiv(n* 64* 64,256),256,0,stream>>>(t2,  t3, n, 128, 128);

    // Coarse-to-fine alignment
    auto launch = [&](const float* sL, const float* dL, const float* prev,
                      float* oOff, float* oDist, int H, int nt, int ntP) {
        int tiles = n * nt * nt;
        align_step<<<cdiv(tiles, WAVES), WAVES * 32, 0, stream>>>(
            sL, dL, prev, oOff, oDist, n, H, H, nt, nt, ntP, ntP);
    };
    launch(s3,  t3,  nullptr, off3, nullptr,  64,  4,  0);
    launch(s2,  t2,  off3,    off2, nullptr, 128,  8,  4);
    launch(s1,  t1,  off2,    off1, nullptr, 256, 16,  8);
    launch(src, dst, off1,    off0, dist0,   512, 32, 16);

    // Per-pixel expansion into d_out
    expand_out<<<cdiv(n*512*512,256),256,0,stream>>>(off0, dist0, out,
                                                     n, 512, 512, 32, 32);
}
